// OutlookAttention_45372034515553
// MI455X (gfx1250) — matrix operand order repaired and checked by reading, not compiled, not run on hardware
//
#include <hip/hip_runtime.h>
#include <math.h>

typedef __attribute__((ext_vector_type(16))) __bf16 v16bf;
typedef __attribute__((ext_vector_type(8)))  __bf16 v8bf;
typedef __attribute__((ext_vector_type(4)))  __bf16 v4bf;
typedef __attribute__((ext_vector_type(2)))  __bf16 v2bf;
typedef __attribute__((ext_vector_type(8)))  float  v8f;
typedef __attribute__((ext_vector_type(4)))  float  f32x4;

#define BM 128
#define BN 128
#define BK 32
#define KDIM 384
#define KTILES (KDIM / BK)

// C[M,N] = A[M,K=384] @ B[K,N] (+bias). mode 0: plain row-major store (ldC).
// mode 1: scatter rows into zero-padded [B,58,58,384] tensor (v-pad layout).
// Double-buffered LDS; bf16 WMMA with f32 accumulation.
__global__ __launch_bounds__(256) void gemm_bf16_wmma(
    const float* __restrict__ A, int ldA,
    const float* __restrict__ B, int ldB, int N,
    const float* __restrict__ bias,
    float* __restrict__ C, int ldC, int mode)
{
  __shared__ __align__(16) __bf16 As[2][BM][BK + 8];   // pitch 40 bf16 = 80B
  __shared__ __align__(16) __bf16 Bt[2][BN][BK + 8];   // transposed (N-major)

  const int tid  = threadIdx.x;
  const int wave = tid >> 5;
  const int lane = tid & 31;
  const int ll   = lane & 15;
  const int hi   = lane >> 4;

  const int m0 = blockIdx.y * BM;
  const int n0 = blockIdx.x * BN;
  const bool fullN = (n0 + BN <= N);

  // staging registers (per-thread slices of the next tile)
  f32x4 aR[4];
  f32x4 bR[4];

  auto stage_load = [&](int k0) {
#pragma unroll
    for (int i = 0; i < 4; ++i) {                  // A: 128x32, 1024 float4
      int idx = tid + i * 256;
      int r   = idx >> 3;                          // 0..127
      int c4  = idx & 7;                           // 0..7
      aR[i] = *(const f32x4*)(A + (size_t)(m0 + r) * ldA + k0 + c4 * 4);
    }
#pragma unroll
    for (int i = 0; i < 2; ++i) {                  // B: 32x128 as 16 row-pairs
      int idx  = tid + i * 256;                    // 0..511
      int krp  = idx >> 5;                         // 0..15 -> rows 2*krp, 2*krp+1
      int c4   = idx & 31;                         // 0..31
      int kr   = krp * 2;
      int coln = n0 + c4 * 4;
      if (fullN) {
        bR[2 * i]     = *(const f32x4*)(B + (size_t)(k0 + kr)     * ldB + coln);
        bR[2 * i + 1] = *(const f32x4*)(B + (size_t)(k0 + kr + 1) * ldB + coln);
      } else {
        const float* s0 = B + (size_t)(k0 + kr)     * ldB + coln;
        const float* s1 = B + (size_t)(k0 + kr + 1) * ldB + coln;
        f32x4 t0 = {}, t1 = {};
#pragma unroll
        for (int j = 0; j < 4; ++j) {
          if (coln + j < N) { t0[j] = s0[j]; t1[j] = s1[j]; }
        }
        bR[2 * i] = t0; bR[2 * i + 1] = t1;
      }
    }
  };

  auto stage_store = [&](int buf) {
#pragma unroll
    for (int i = 0; i < 4; ++i) {
      int idx = tid + i * 256;
      int r   = idx >> 3;
      int c4  = idx & 7;
      v4bf p = { (__bf16)aR[i].x, (__bf16)aR[i].y, (__bf16)aR[i].z, (__bf16)aR[i].w };
      *(v4bf*)(&As[buf][r][c4 * 4]) = p;
    }
#pragma unroll
    for (int i = 0; i < 2; ++i) {
      int idx = tid + i * 256;
      int krp = idx >> 5;
      int c4  = idx & 31;
      int kr  = krp * 2;
#pragma unroll
      for (int j = 0; j < 4; ++j) {
        v2bf p = { (__bf16)bR[2 * i][j], (__bf16)bR[2 * i + 1][j] };
        *(v2bf*)(&Bt[buf][c4 * 4 + j][kr]) = p;   // K-adjacent pair packed b32
      }
    }
  };

  v8f acc[8] = {};

  stage_load(0);
  stage_store(0);
  __syncthreads();

  for (int kt = 0; kt < KTILES; ++kt) {
    const int buf = kt & 1;
    if (kt + 1 < KTILES) stage_load((kt + 1) * BK);   // overlap with compute

    // A fragment: lane<16 row M=ll, e0-7 -> K base..base+7, e8-15 -> K base+16.. (base=8*hi)
    const int arow  = wave * 16 + ll;
    const int abase = hi * 8;
    union uf { v16bf v; v8bf h[2]; };
    uf af;
    af.h[0] = *(const v8bf*)(&As[buf][arow][abase]);
    af.h[1] = *(const v8bf*)(&As[buf][arow][abase + 16]);

    const int bbase = hi * 8;     // B frag: col N=ll, e0-7 -> K base..base+7, e8-15 -> K base+16..
#pragma unroll
    for (int g = 0; g < 2; ++g) {
      uf bfr[4];
#pragma unroll
      for (int u = 0; u < 4; ++u) {
        int bcol = (g * 4 + u) * 16 + ll;
        bfr[u].h[0] = *(const v8bf*)(&Bt[buf][bcol][bbase]);
        bfr[u].h[1] = *(const v8bf*)(&Bt[buf][bcol][bbase + 16]);
      }
#pragma unroll
      for (int u = 0; u < 4; ++u) {
        acc[g * 4 + u] = __builtin_amdgcn_wmma_f32_16x16x32_bf16(
            false, af.v, false, bfr[u].v, (short)0, acc[g * 4 + u], false, false);
      }
    }

    if (kt + 1 < KTILES) {
      stage_store(buf ^ 1);
      __syncthreads();
    }
  }

  // ---- epilogue: C layout VGPR r -> M = r + 8*hi, N = ll ----
#pragma unroll
  for (int ni = 0; ni < 8; ++ni) {
    int gn = n0 + ni * 16 + ll;
    if (gn >= N) continue;
    float bv = bias ? bias[gn] : 0.f;
#pragma unroll
    for (int r = 0; r < 8; ++r) {
      int gm = m0 + wave * 16 + r + hi * 8;
      float val = acc[ni][r] + bv;
      if (mode == 0) {
        C[(size_t)gm * ldC + gn] = val;
      } else {
        int b   = gm / 3136;
        int rem = gm - b * 3136;
        int y   = rem / 56;
        int x   = rem - y * 56;
        C[(((size_t)(b * 58 + y + 1)) * 58 + (x + 1)) * 384 + gn] = val;
      }
    }
  }
}

// 2x2 average pool: x[32,56,56,384] -> xp[32,28,28,384]
__global__ __launch_bounds__(256) void pool_kernel(
    const float* __restrict__ x, float* __restrict__ xp, int total)
{
  int idx = blockIdx.x * 256 + threadIdx.x;
  if (idx >= total) return;
  int c = idx % 384;
  int s = idx / 384;
  int j = s % 28; s /= 28;
  int i = s % 28;
  int b = s / 28;
  const float* base = x + (((size_t)(b * 56 + 2 * i)) * 56 + 2 * j) * 384 + c;
  float v = base[0] + base[384] + base[56 * 384] + base[56 * 384 + 384];
  xp[idx] = v * 0.25f;
}

// softmax over k (9) + out[q,d] = sum_k a[q,k] * v[k,d] per (b,p) location
__global__ __launch_bounds__(256) void attn_kernel(
    const float* __restrict__ a, const float* __restrict__ vpad,
    float* __restrict__ out)
{
  __shared__ float sA[108][10];     // 12 heads * 9 q rows, 9 probs (+pad)
  __shared__ __align__(16) float sV[9][384];

  int bp = blockIdx.x;              // b*784 + p
  int b  = bp / 784;
  int p  = bp - b * 784;
  int pi = p / 28, pj = p - pi * 28;
  int tid = threadIdx.x;

  if (tid < 108) {                  // one (head,q) row each
    const float* row = a + (size_t)bp * 972 + tid * 9;   // head*81 + q*9
    float r[9];
    float m = -1e30f;
#pragma unroll
    for (int k = 0; k < 9; ++k) { r[k] = row[k]; m = fmaxf(m, r[k]); }
    const float scale = 0.1767766952966369f;             // 1/sqrt(32)
    float s = 0.f;
#pragma unroll
    for (int k = 0; k < 9; ++k) { r[k] = __expf((r[k] - m) * scale); s += r[k]; }
    float inv = 1.f / s;
#pragma unroll
    for (int k = 0; k < 9; ++k) sA[tid][k] = r[k] * inv;
  }

  // gather 3x3 patch of v from padded layout (rows 2i+ki, cols 2j+kj)
  for (int idx = tid; idx < 864; idx += 256) {           // 864 float4 = 9*384 floats
    int k  = idx / 96;
    int c4 = idx - k * 96;
    int ki = k / 3, kj = k - ki * 3;
    const f32x4* src = (const f32x4*)(vpad +
        (((size_t)(b * 58 + 2 * pi + ki)) * 58 + (2 * pj + kj)) * 384) + c4;
    *((f32x4*)&sV[k][c4 * 4]) = *src;
  }
  __syncthreads();

  float* obase = out + (size_t)bp * 9 * 384;             // [q][head*32+d]
  for (int o = tid; o < 3456; o += 256) {
    int head = o / 288;             // 9*32
    int rem  = o - head * 288;
    int q    = rem / 32;
    int d    = rem - q * 32;
    int c    = head * 32 + d;
    const float* arow = sA[head * 9 + q];
    float acc = 0.f;
#pragma unroll
    for (int k = 0; k < 9; ++k) acc += arow[k] * sV[k][c];
    obase[q * 384 + c] = acc;
  }
}

// deterministic fold: gather the <=4 (qi,qj) contributions per output pixel
__global__ __launch_bounds__(256) void fold_kernel(
    const float* __restrict__ out, float* __restrict__ folded, int total)
{
  int idx = blockIdx.x * 256 + threadIdx.x;
  if (idx >= total) return;
  int c = idx % 384;
  int s = idx / 384;
  int x = s % 56; s /= 56;
  int y = s % 56;
  int b = s / 56;
  int yp = y + 1, xp = x + 1;       // padded coords
  float acc = 0.f;
  for (int qi = 0; qi < 3; ++qi) {
    int ty = yp - qi;
    if (ty < 0 || (ty & 1) || ty >= 56) continue;
    int pi = ty >> 1;
    for (int qj = 0; qj < 3; ++qj) {
      int tx = xp - qj;
      if (tx < 0 || (tx & 1) || tx >= 56) continue;
      int pj = tx >> 1;
      int p = pi * 28 + pj;
      int q = qi * 3 + qj;
      acc += out[(((size_t)(b * 784 + p)) * 9 + q) * 384 + c];
    }
  }
  folded[idx] = acc;
}

extern "C" void kernel_launch(void* const* d_in, const int* in_sizes, int n_in,
                              void* d_out, int out_size, void* d_ws, size_t ws_size,
                              hipStream_t stream) {
  const float* x   = (const float*)d_in[0];
  const float* W_v = (const float*)d_in[1];
  const float* W_a = (const float*)d_in[2];
  const float* b_a = (const float*)d_in[3];
  const float* W_o = (const float*)d_in[4];
  const float* b_o = (const float*)d_in[5];
  float* y = (float*)d_out;

  char* ws = (char*)d_ws;
  size_t off = 0;
  float* vpad   = (float*)(ws + off); off += (size_t)32 * 58 * 58 * 384 * 4;   // 165.3 MB
  float* xpool  = (float*)(ws + off); off += (size_t)32 * 784 * 384 * 4;       //  38.5 MB
  float* amat   = (float*)(ws + off); off += (size_t)32 * 784 * 972 * 4;       //  97.5 MB
  float* outb   = (float*)(ws + off); off += (size_t)32 * 784 * 9 * 384 * 4;   // 346.8 MB
  float* folded = (float*)(ws + off); off += (size_t)32 * 3136 * 384 * 4;      // 154.1 MB

  // zero padded v buffer (borders must be zero for patch gather)
  hipMemsetAsync(vpad, 0, (size_t)32 * 58 * 58 * 384 * 4, stream);

  // v = x @ W_v, scattered into padded layout
  gemm_bf16_wmma<<<dim3(3, 784), 256, 0, stream>>>(x, 384, W_v, 384, 384,
                                                   nullptr, vpad, 0, 1);
  // 2x2 average pool
  int ptotal = 32 * 784 * 384;
  pool_kernel<<<(ptotal + 255) / 256, 256, 0, stream>>>(x, xpool, ptotal);

  // a = xpool @ W_a + b_a   (N=972, guarded)
  gemm_bf16_wmma<<<dim3(8, 196), 256, 0, stream>>>(xpool, 384, W_a, 972, 972,
                                                   b_a, amat, 972, 0);
  // softmax + attention apply
  attn_kernel<<<25088, 256, 0, stream>>>(amat, vpad, outb);

  // fold (gather) back to [32,56,56,384]
  int ftotal = 32 * 3136 * 384;
  fold_kernel<<<(ftotal + 255) / 256, 256, 0, stream>>>(outb, folded, ftotal);

  // y = folded @ W_o + b_o
  gemm_bf16_wmma<<<dim3(3, 784), 256, 0, stream>>>(folded, 384, W_o, 384, 384,
                                                   b_o, y, 384, 0);
}
